// MSDeformableAttention_38938173505550
// MI455X (gfx1250) — compile-verified, matrix operand order checked by reading, not run on hardware
//
#include <hip/hip_runtime.h>
#include <hip/hip_bf16.h>

typedef __attribute__((ext_vector_type(16))) _Float16 v16h;
typedef __attribute__((ext_vector_type(8)))  float    v8f;

#define BS      16
#define LQ      300
#define CDIM    256
#define NHEADS  8
#define HD      32
#define SUMP    16
#define NCOLS   384   // 256 offset cols + 128 attn cols
#define NROWS   (BS * LQ)   // 4800
#define LV      8500

// ---------------------------------------------------------------------------
// Templated WMMA K-loop: LDW is compile-time so all 128 B-load offsets fold
// into the global_load immediate offset field (single base VGPR pair).
// ---------------------------------------------------------------------------
template <int LDW>
__device__ __forceinline__ v8f gemm_ktile(const _Float16* __restrict__ sArow,
                                          const float*    __restrict__ Wlane,
                                          int khi)
{
    v8f acc = {};
    #pragma unroll
    for (int kk = 0; kk < CDIM; kk += 32) {
        v16h a, b;
        #pragma unroll
        for (int i = 0; i < 16; ++i) {
            // ISA 16-bit A/B 16x32 layout: K = (i&7) + ((i>>3)<<4) + 8*(lane>=16)
            const int Kloc = (i & 7) + ((i >> 3) << 4);
            a[i] = sArow[kk + Kloc + khi];
            b[i] = (_Float16)Wlane[(kk + Kloc) * LDW];   // compile-time offset
        }
        acc = __builtin_amdgcn_wmma_f32_16x16x32_f16(
                  /*neg_a=*/false, a, /*neg_b=*/false, b,
                  /*c_mod=*/(short)0, acc, /*reuse_a=*/false, /*reuse_b=*/false);
    }
    return acc;
}

// ---------------------------------------------------------------------------
// Kernel A: proj[4800, 384] = f16wmma( query[4800,256] @ [w_off | w_attn] ) + bias
// grid (3, 300), block 256 (8 waves). Each wave -> one 16x16 tile of D.
// ---------------------------------------------------------------------------
__global__ __launch_bounds__(256) void msda_proj_wmma(
    const float* __restrict__ q,
    const float* __restrict__ w_off,  const float* __restrict__ b_off,
    const float* __restrict__ w_attn, const float* __restrict__ b_attn,
    float* __restrict__ proj)
{
    __shared__ _Float16 sA[16 * CDIM];   // 16-row query tile, f16 (8 KB)

    const int m0   = blockIdx.y * 16;
    const int wave = threadIdx.x >> 5;
    const int lane = threadIdx.x & 31;
    const int n0   = blockIdx.x * 128 + wave * 16;

    // Cooperative A-tile load (coalesced), convert f32 -> f16 once for 8 waves.
    for (int idx = threadIdx.x; idx < 16 * CDIM; idx += 256) {
        int r = idx >> 8, k = idx & 255;
        sA[idx] = (_Float16)q[(size_t)(m0 + r) * CDIM + k];
    }
    __syncthreads();

    const int mrow = lane & 15;           // A: M index held by this lane
    const int col  = n0 + (lane & 15);    // B/D: N index held by this lane
    const int khi  = (lane >> 4) << 3;    // K sub-block offset per half-wave
    const _Float16* sArow = &sA[mrow * CDIM];

    // n0 (hence col<256) is wave-uniform: pick weight matrix once, then run a
    // constant-stride instantiation of the K-loop.
    v8f   acc;
    float bias;
    if (n0 < 256) {
        bias = b_off[col];
        acc  = gemm_ktile<256>(sArow, w_off + col + khi * 256, khi);
    } else {
        bias = b_attn[col - 256];
        acc  = gemm_ktile<128>(sArow, w_attn + (col - 256) + khi * 128, khi);
    }

    const int mbase = m0 + ((lane >> 4) << 3);   // D: lanes16-31 hold M=8..15
    #pragma unroll
    for (int r = 0; r < 8; ++r)
        proj[(size_t)(mbase + r) * NCOLS + col] = acc[r] + bias;
}

// ---------------------------------------------------------------------------
// Kernel B: softmax + bilinear sampling + weighted sum.
// 4800 blocks (one per (b,q)), 256 threads = 8 heads x 32 channels.
// ---------------------------------------------------------------------------
__global__ __launch_bounds__(256) void msda_sample(
    const float* __restrict__ proj,
    const float* __restrict__ ref,     // (bs, Lq, 1, 4)
    const float* __restrict__ value,   // (bs, Lv, 256)
    float* __restrict__ out)           // (bs, Lq, 256)
{
    __shared__ float s_proj[NCOLS];
    __shared__ float s_ref[4];

    const int row = blockIdx.x;            // b*LQ + q
    const int b   = row / LQ;

    for (int idx = threadIdx.x; idx < NCOLS; idx += 256)
        s_proj[idx] = proj[(size_t)row * NCOLS + idx];
    if (threadIdx.x < 4)
        s_ref[threadIdx.x] = ref[(size_t)row * 4 + threadIdx.x];
    __syncthreads();

    const int head = threadIdx.x >> 5;
    const int c    = threadIdx.x & 31;

    // ---- per-head softmax over 16 points (registers; broadcast LDS reads) ----
    float w[SUMP];
    float mx = -1e30f;
    #pragma unroll
    for (int p = 0; p < SUMP; ++p) {
        w[p] = s_proj[256 + head * SUMP + p];
        mx = fmaxf(mx, w[p]);
    }
    float ssum = 0.f;
    #pragma unroll
    for (int p = 0; p < SUMP; ++p) {
        w[p] = __expf(w[p] - mx);
        ssum += w[p];
    }
    const float inv = __frcp_rn(ssum);

    const float cx  = s_ref[0], cy = s_ref[1];
    // loc = center + off * (1/n_pts=0.25) * wh * (OFFSET_SCALE=0.5)
    const float swx = s_ref[2] * 0.125f, swy = s_ref[3] * 0.125f;

    const int   vstart[4] = {0, 6400, 8000, 8400};
    const int   ldim[4]   = {80, 40, 20, 10};

    const size_t vbase = (size_t)b * LV;
    const int    ch    = head * HD + c;

    float acc = 0.f;
    #pragma unroll
    for (int p = 0; p < SUMP; ++p) {
        const int l  = p >> 2;
        const int Wl = ldim[l], Hl = ldim[l];
        const int vs = vstart[l];

        const float offx = s_proj[head * (SUMP * 2) + p * 2 + 0];
        const float offy = s_proj[head * (SUMP * 2) + p * 2 + 1];
        const float locx = cx + offx * swx;
        const float locy = cy + offy * swy;

        // grid = 2*loc - 1 ; x = (grid+1)*W/2 - 0.5 = loc*W - 0.5
        const float x = locx * (float)Wl - 0.5f;
        const float y = locy * (float)Hl - 0.5f;
        const float xf = floorf(x), yf = floorf(y);
        const int   x0 = (int)xf,  y0 = (int)yf;
        const float fx = x - xf,   fy = y - yf;

        float s = 0.f;
        #pragma unroll
        for (int t = 0; t < 4; ++t) {
            const int   ix  = x0 + (t & 1);
            const int   iy  = y0 + (t >> 1);
            const float wx  = (t & 1)  ? fx : (1.f - fx);
            const float wy  = (t >> 1) ? fy : (1.f - fy);
            if (ix >= 0 && ix < Wl && iy >= 0 && iy < Hl) {
                const float v = value[(vbase + vs + (size_t)iy * Wl + ix) * CDIM + ch];
                s = fmaf(wx * wy, v, s);
            }
        }
        acc = fmaf(w[p] * inv, s, acc);
    }

    out[(size_t)row * CDIM + ch] = acc;
}

// ---------------------------------------------------------------------------
extern "C" void kernel_launch(void* const* d_in, const int* in_sizes, int n_in,
                              void* d_out, int out_size, void* d_ws, size_t ws_size,
                              hipStream_t stream) {
    const float* query  = (const float*)d_in[0];
    const float* refpts = (const float*)d_in[1];
    const float* value  = (const float*)d_in[2];
    const float* w_off  = (const float*)d_in[3];
    const float* b_off  = (const float*)d_in[4];
    const float* w_attn = (const float*)d_in[5];
    const float* b_attn = (const float*)d_in[6];
    float*       out    = (float*)d_out;
    float*       proj   = (float*)d_ws;   // 4800 * 384 * 4B = 7.37 MB

    dim3 gridA(NCOLS / 128, NROWS / 16);  // (3, 300)
    msda_proj_wmma<<<gridA, 256, 0, stream>>>(query, w_off, b_off, w_attn, b_attn, proj);

    msda_sample<<<NROWS, 256, 0, stream>>>(proj, refpts, value, out);
}
